// MoENRX_6794638262744
// MI455X (gfx1250) — matrix operand order repaired and checked by reading, not compiled, not run on hardware
//
#include <hip/hip_runtime.h>
#include <hip/hip_bf16.h>

typedef __attribute__((ext_vector_type(16))) _Float16 v16h;
typedef __attribute__((ext_vector_type(8)))  _Float16 v8h;
typedef __attribute__((ext_vector_type(8)))  float    v8f;

#define PTOT   114688          // 64 * 128 * 14 pixels
#define SC     128
#define SY     14
#define PIXB   (SC*SY)         // 1792 pixels per batch sample

// ---------------------------------------------------------------------------
// CDNA5 async helpers
// ---------------------------------------------------------------------------
__device__ __forceinline__ void async_b128(unsigned ldsOff, const void* gaddr)
{
    // per-lane: LDS[ldsOff] <= 16 bytes from gaddr, tracked with ASYNCcnt
    asm volatile("global_load_async_to_lds_b128 %0, %1, off"
                 :: "v"(ldsOff), "v"(gaddr) : "memory");
}
__device__ __forceinline__ void wait_async0()
{
#if __has_builtin(__builtin_amdgcn_s_wait_asynccnt)
    __builtin_amdgcn_s_wait_asynccnt(0);
#else
    asm volatile("s_wait_asynccnt 0x0" ::: "memory");
#endif
}

// ---------------------------------------------------------------------------
// Weight repack: OIHW fp32 -> [np][K] f16, K = ks*ks*cis, zero-padded
// k = (ky*ks+kx)*cis + ci ; rows co>=cout and cols ci>=cin are zero.
// ---------------------------------------------------------------------------
__global__ void repack_kernel(const float* __restrict__ w, _Float16* __restrict__ wt,
                              int cout, int cin, int cis, int ks, int np)
{
    int K = ks * ks * cis;
    size_t n = (size_t)np * K;
    size_t idx = (size_t)blockIdx.x * 256 + threadIdx.x;
    if (idx >= n) return;
    int co  = (int)(idx / K);
    int k   = (int)(idx % K);
    int tap = k / cis;
    int ci  = k - tap * cis;
    int ky  = tap / ks, kx = tap % ks;
    _Float16 v = (_Float16)0.f;
    if (co < cout && ci < cin)
        v = (_Float16)w[(((size_t)co * cin + ci) * ks + ky) * ks + kx];
    wt[idx] = v;
}

// ---------------------------------------------------------------------------
// Per-sample normalization + pilot features -> [p][32] f16 (ch 18..31 zero)
// ---------------------------------------------------------------------------
__global__ __launch_bounds__(256) void norm_prep(const float* __restrict__ x,
                                                 _Float16* __restrict__ in32)
{
    int b = blockIdx.x, tid = threadIdx.x;
    __shared__ float sm[256], sq[256];
    const float* xb = x + (size_t)b * 16 * PIXB;
    float s = 0.f, s2 = 0.f;
    for (int i = tid; i < 16 * PIXB; i += 256) { float v = xb[i]; s += v; s2 += v * v; }
    sm[tid] = s; sq[tid] = s2; __syncthreads();
    for (int st = 128; st > 0; st >>= 1) {
        if (tid < st) { sm[tid] += sm[tid + st]; sq[tid] += sq[tid + st]; }
        __syncthreads();
    }
    const float inv_n = 1.f / (16.f * (float)PIXB);
    float mu  = sm[0] * inv_n;
    float var = sq[0] * inv_n - mu * mu;
    float sd  = sqrtf(var > 0.f ? var : 0.f);
    float inv = 1.f / (sd + 1e-6f);
    for (int i = tid; i < PIXB; i += 256) {
        int ss = i / SY, tt = i % SY;
        _Float16* o = in32 + ((size_t)b * PIXB + i) * 32;
#pragma unroll
        for (int c = 0; c < 16; ++c)
            o[c] = (_Float16)((xb[(size_t)c * PIXB + i] - mu) * inv);
        float df = (ss & 1) ? (1.f / (float)SC) : 0.f;
        float dt = fminf(fabsf((float)tt - 2.f), fabsf((float)tt - 11.f)) / (float)SY;
        o[16] = (_Float16)df;
        o[17] = (_Float16)dt;
#pragma unroll
        for (int c = 18; c < 32; ++c) o[c] = (_Float16)0.f;
    }
}

// ---------------------------------------------------------------------------
// Implicit-GEMM conv via WMMA f32 += f16 x f16, async-to-LDS double buffered.
// Block tile M128 x N64, 8 waves (4M x 2N), wave tile 32M x 32N = 4 acc.
// Strides padded to 32 -> every 32-K chunk is one contiguous 64B run per row.
// ---------------------------------------------------------------------------
__device__ __forceinline__ v16h ld_frag(const _Float16* p0, const _Float16* p1)
{
    v8h a = *(const v8h*)p0, b = *(const v8h*)p1;
    v16h r;
#pragma unroll
    for (int i = 0; i < 8; ++i) { r[i] = a[i]; r[i + 8] = b[i]; }
    return r;
}

__global__ __launch_bounds__(256)
void conv_wmma(const _Float16* __restrict__ in, const _Float16* __restrict__ wt,
               const float* __restrict__ bias, const _Float16* res,
               _Float16* out16, float* out32,
               int Cis, int Cout, int Cos, int ks, int relu)
{
    __shared__ alignas(16) _Float16 lA[2][128][40];  // 80B row pitch: conflict-free b128
    __shared__ alignas(16) _Float16 lB[2][64][40];

    const int Ktot = ks * ks * Cis;                   // multiple of 32 by construction
    const int cpt  = Cis >> 5;                        // K-chunks per tap
    const int tid  = threadIdx.x;
    const int lane = tid & 31;
    const int wv   = tid >> 5;
    const int wM   = wv & 3;                          // 4 waves x 32 rows
    const int wN   = wv >> 2;                         // 2 waves x 32 cols
    const int pBase = blockIdx.x * 128;
    const int nBase = blockIdx.y * 64;

    // A fill: thread -> (row, 16-f16 half-row);  B fill: thread -> (row, 8-f16 seg)
    const int fRowA = tid >> 1;          // 0..127
    const int fKA   = (tid & 1) * 16;    // 0 or 16
    const int fRowB = tid >> 2;          // 0..63
    const int fKB   = (tid & 3) * 8;

    const int pA  = pBase + fRowA;
    const int tA  = pA % SY;
    const int sA  = (pA / SY) % SC;
    const int pRowBase = (pA / PIXB) * PIXB;
    const int ngFill = nBase + fRowB;

    auto stage = [&](int kc, int buf) {
        // weights: always valid (padded rows/cols)
        async_b128((unsigned)(size_t)&lB[buf][fRowB][fKB],
                   wt + (size_t)ngFill * Ktot + (kc << 5) + fKB);
        // activations: one tap per chunk
        int tap = 0, ci = (kc << 5) + fKA, dy = 0, dx = 0;
        if (ks == 3) {
            tap = kc / cpt;
            ci  = ((kc - tap * cpt) << 5) + fKA;
            dy  = tap / 3 - 1;
            dx  = tap % 3 - 1;
        }
        int ss = sA + dy, tt = tA + dx;
        if (ss >= 0 && ss < SC && tt >= 0 && tt < SY) {
            const _Float16* ga = in + (size_t)(pRowBase + ss * SY + tt) * Cis + ci;
            async_b128((unsigned)(size_t)&lA[buf][fRowA][fKA],     ga);
            async_b128((unsigned)(size_t)&lA[buf][fRowA][fKA + 8], ga + 8);
        } else {
            v8h z = {};
            *(v8h*)&lA[buf][fRowA][fKA]     = z;     // SAME-pad boundary row
            *(v8h*)&lA[buf][fRowA][fKA + 8] = z;
        }
    };

    v8f acc00 = {}, acc01 = {}, acc10 = {}, acc11 = {};
    const int nChunks = Ktot >> 5;

    stage(0, 0);
    for (int kc = 0; kc < nChunks; ++kc) {
        const int buf = kc & 1;
        wait_async0();                                // own async writes landed
        __syncthreads();                              // publish LDS across waves
        if (kc + 1 < nChunks) stage(kc + 1, buf ^ 1); // prefetch next chunk

        const int r  = lane & 15;
        const int hi = lane >> 4;
        const int ak = hi ? 8 : 0;
        v16h af0 = ld_frag(&lA[buf][wM * 32 + r][ak],      &lA[buf][wM * 32 + r][ak + 16]);
        v16h af1 = ld_frag(&lA[buf][wM * 32 + 16 + r][ak], &lA[buf][wM * 32 + 16 + r][ak + 16]);
        const int bk = hi ? 8 : 0;
        v16h bf0 = ld_frag(&lB[buf][wN * 32 + r][bk],      &lB[buf][wN * 32 + r][bk + 16]);
        v16h bf1 = ld_frag(&lB[buf][wN * 32 + 16 + r][bk], &lB[buf][wN * 32 + 16 + r][bk + 16]);

        acc00 = __builtin_amdgcn_wmma_f32_16x16x32_f16(false, af0, false, bf0, (short)0, acc00, false, false);
        acc01 = __builtin_amdgcn_wmma_f32_16x16x32_f16(false, af0, false, bf1, (short)0, acc01, false, false);
        acc10 = __builtin_amdgcn_wmma_f32_16x16x32_f16(false, af1, false, bf0, (short)0, acc10, false, false);
        acc11 = __builtin_amdgcn_wmma_f32_16x16x32_f16(false, af1, false, bf1, (short)0, acc11, false, false);
        __syncthreads();                              // all reads done before rewrite
    }

    // epilogue: bias + relu + residual; f16 (padded stride) / f32 (dense) stores
    const int nCol = lane & 15;
    const int mAdd = (lane >> 4) * 8;
#pragma unroll
    for (int msub = 0; msub < 2; ++msub) {
#pragma unroll
        for (int rr = 0; rr < 8; ++rr) {
            int p = pBase + wM * 32 + msub * 16 + rr + mAdd;
#pragma unroll
            for (int half = 0; half < 2; ++half) {
                int ng = nBase + wN * 32 + half * 16 + nCol;
                if (ng < Cout) {
                    float v = (msub ? (half ? acc11[rr] : acc10[rr])
                                    : (half ? acc01[rr] : acc00[rr])) + bias[ng];
                    if (relu) v = v > 0.f ? v : 0.f;
                    if (res)   v += (float)res[(size_t)p * Cos + ng];
                    if (out16) out16[(size_t)p * Cos + ng] = (_Float16)v;
                    if (out32) out32[(size_t)p * Cout + ng] = v;
                }
            }
        }
    }
}

// ---------------------------------------------------------------------------
// Router feature pooling: per (b,c) mean & max over pixels (stride-64 input)
// ---------------------------------------------------------------------------
__global__ void feats_kernel(const _Float16* __restrict__ sh, float* __restrict__ feats)
{
    int b = blockIdx.x, tid = threadIdx.x;
    if (tid < 56) {
        const _Float16* sp = sh + (size_t)b * PIXB * 64 + tid;
        float s = 0.f, mx = -3.4e38f;
        for (int i = 0; i < PIXB; ++i) { float v = (float)sp[(size_t)i * 64]; s += v; mx = v > mx ? v : mx; }
        feats[b * 112 + tid]      = s / (float)PIXB;
        feats[b * 112 + 56 + tid] = mx;
    }
}

// ---------------------------------------------------------------------------
// Router MLP + softmax + argmax one-hot
// ---------------------------------------------------------------------------
__global__ void router_kernel(const float* __restrict__ feats,
                              const float* __restrict__ rw1, const float* __restrict__ rb1,
                              const float* __restrict__ rw2, const float* __restrict__ rb2,
                              float* out_logits, float* out_probs, float* out_w, int* sel)
{
    int b = blockIdx.x, tid = threadIdx.x;   // 64 threads
    __shared__ float hid[64];
    const float* f = feats + b * 112;
    float a = rb1[tid];
    for (int i = 0; i < 112; ++i) a += f[i] * rw1[tid * 112 + i];
    hid[tid] = a > 0.f ? a : 0.f;
    __syncthreads();
    if (tid == 0) {
        float lg[3];
        for (int j = 0; j < 3; ++j) {
            float s = rb2[j];
            for (int i = 0; i < 64; ++i) s += hid[i] * rw2[j * 64 + i];
            lg[j] = s;
        }
        float mx = fmaxf(lg[0], fmaxf(lg[1], lg[2]));
        float e0 = expf(lg[0] - mx), e1 = expf(lg[1] - mx), e2 = expf(lg[2] - mx);
        float den = e0 + e1 + e2;
        int am = 0; float bv = lg[0];
        if (lg[1] > bv) { bv = lg[1]; am = 1; }
        if (lg[2] > bv) { bv = lg[2]; am = 2; }
        out_logits[b * 3 + 0] = lg[0]; out_logits[b * 3 + 1] = lg[1]; out_logits[b * 3 + 2] = lg[2];
        out_probs[b * 3 + 0] = e0 / den; out_probs[b * 3 + 1] = e1 / den; out_probs[b * 3 + 2] = e2 / den;
        out_w[b * 3 + 0] = (am == 0) ? 1.f : 0.f;
        out_w[b * 3 + 1] = (am == 1) ? 1.f : 0.f;
        out_w[b * 3 + 2] = (am == 2) ? 1.f : 0.f;
        sel[b] = am;
    }
}

// ---------------------------------------------------------------------------
// One-hot weighted sum == select argmax expert; scatter into NCHW d_out
// ---------------------------------------------------------------------------
__global__ void combine_kernel(const float* __restrict__ llr, const float* __restrict__ chan,
                               const int* __restrict__ sel,
                               float* __restrict__ outL, float* __restrict__ outC)
{
    size_t idx = (size_t)blockIdx.x * 256 + threadIdx.x;     // B * 12 * PIXB
    if (idx >= (size_t)64 * 12 * PIXB) return;
    int i = (int)(idx % PIXB);
    int c = (int)((idx / PIXB) % 12);
    int b = (int)(idx / ((size_t)PIXB * 12));
    int e = sel[b];
    size_t p = (size_t)b * PIXB + i;
    if (c < 4)
        outL[((size_t)b * 4 + c) * PIXB + i] = llr[((size_t)e * PTOT + p) * 4 + c];
    else
        outC[((size_t)b * 8 + (c - 4)) * PIXB + i] = chan[((size_t)e * PTOT + p) * 8 + (c - 4)];
}

// ---------------------------------------------------------------------------
// Host orchestration
// ---------------------------------------------------------------------------
extern "C" void kernel_launch(void* const* d_in, const int* in_sizes, int n_in,
                              void* d_out, int out_size, void* d_ws, size_t ws_size,
                              hipStream_t stream)
{
    (void)in_sizes; (void)n_in; (void)out_size; (void)ws_size;

    char* wsBase = (char*)d_ws;
    size_t off = 0;
    auto alloc = [&](size_t bytes) -> void* {
        void* p = wsBase + off;
        off += bytes; off = (off + 255) & ~(size_t)255;
        return p;
    };
    _Float16* wbuf  = (_Float16*)alloc((size_t)4 << 20);           // padded f16 weights
    _Float16* in32  = (_Float16*)alloc((size_t)PTOT * 32 * 2);
    _Float16* bufA  = (_Float16*)alloc((size_t)PTOT * 64 * 2);     // stem a1 / expert t1
    _Float16* bufB  = (_Float16*)alloc((size_t)PTOT * 64 * 2);     // stem a2 / expert t2
    _Float16* shbuf = (_Float16*)alloc((size_t)PTOT * 64 * 2);     // shared feats (56 used)
    _Float16* hbuf  = (_Float16*)alloc((size_t)PTOT * 64 * 2);     // expert state (56 used)
    float*    llrAll  = (float*)alloc((size_t)3 * PTOT * 4 * 4);
    float*    chanAll = (float*)alloc((size_t)3 * PTOT * 8 * 4);
    float*    feats   = (float*)alloc((size_t)64 * 112 * 4);
    int*      sel     = (int*)alloc((size_t)64 * 4);
    _Float16* robuf = bufA;   // readout scratch overlays bufA+bufB (29.3MB contiguous)

    float* outLogits = (float*)d_out;
    float* outChan   = outLogits + 458752;
    float* outRLog   = outLogits + 1376256;
    float* outRProb  = outLogits + 1376448;
    float* outRW     = outLogits + 1376640;

    size_t wpos = 0;
    auto repack = [&](int widx, int cin, int cis, int cout, int ks) -> size_t {
        size_t o = wpos;
        int np = ((cout + 63) / 64) * 64;
        size_t n = (size_t)np * ks * ks * cis;
        repack_kernel<<<dim3((unsigned)((n + 255) / 256)), 256, 0, stream>>>(
            (const float*)d_in[widx], wbuf + o, cout, cin, cis, ks, np);
        wpos += n; wpos = (wpos + 63) & ~(size_t)63;   // keep rows 128B aligned
        return o;
    };
    auto conv = [&](const _Float16* in, size_t wOff, int bIdx, const _Float16* res,
                    _Float16* o16, float* o32, int cis, int cout, int cos, int ks, int relu) {
        dim3 g(PTOT / 128, (unsigned)((cout + 63) / 64));
        conv_wmma<<<g, 256, 0, stream>>>(in, wbuf + wOff, (const float*)d_in[bIdx],
                                         res, o16, o32, cis, cout, cos, ks, relu);
    };

    const int hids[3] = {32, 48, 64};
    const int ros[3]  = {96, 128, 128};

    size_t stemW[3];
    stemW[0] = repack(1, 18, 32, 64, 3);
    stemW[1] = repack(3, 64, 64, 64, 3);
    stemW[2] = repack(5, 64, 64, 56, 3);
    size_t expW[3][8][3], roW[3][4];
    for (int e = 0; e < 3; ++e) {
        int base = 11 + e * 56, hid = hids[e];
        int hs = (hid + 31) & ~31;                       // hidden stride
        int ro = ros[e];
        for (int blk = 0; blk < 8; ++blk) {
            int wb = base + blk * 6;
            expW[e][blk][0] = repack(wb + 0, 56, 64, hid, 1);
            expW[e][blk][1] = repack(wb + 2, hid, hs, hid, 3);
            expW[e][blk][2] = repack(wb + 4, hid, hs, 56, 1);
        }
        roW[e][0] = repack(base + 48, 56, 64, ro, 1);
        roW[e][1] = repack(base + 50, ro, ro, 4, 1);
        roW[e][2] = repack(base + 52, 56, 64, ro, 1);
        roW[e][3] = repack(base + 54, ro, ro, 8, 1);
    }

    norm_prep<<<64, 256, 0, stream>>>((const float*)d_in[0], in32);

    // stem
    conv(in32, stemW[0], 2, nullptr, bufA,  nullptr, 32, 64, 64, 3, 1);
    conv(bufA, stemW[1], 4, nullptr, bufB,  nullptr, 64, 64, 64, 3, 1);
    conv(bufB, stemW[2], 6, nullptr, shbuf, nullptr, 64, 56, 64, 3, 0);

    // router
    feats_kernel<<<64, 64, 0, stream>>>(shbuf, feats);
    router_kernel<<<64, 64, 0, stream>>>(feats,
        (const float*)d_in[7], (const float*)d_in[8],
        (const float*)d_in[9], (const float*)d_in[10],
        outRLog, outRProb, outRW, sel);

    // experts
    for (int e = 0; e < 3; ++e) {
        int base = 11 + e * 56, hid = hids[e];
        int hs = (hid + 31) & ~31;
        int ro = ros[e];
        for (int blk = 0; blk < 8; ++blk) {
            int wb = base + blk * 6;
            const _Float16* hin = (blk == 0) ? shbuf : hbuf;
            conv(hin,  expW[e][blk][0], wb + 1, nullptr, bufA, nullptr, 64, hid, hs, 1, 1);
            conv(bufA, expW[e][blk][1], wb + 3, nullptr, bufB, nullptr, hs,  hid, hs, 3, 1);
            conv(bufB, expW[e][blk][2], wb + 5, hin,     hbuf, nullptr, hs,  56,  64, 1, 0);
        }
        conv(hbuf,  roW[e][0], base + 49, nullptr, robuf, nullptr, 64, ro, ro, 1, 1);
        conv(robuf, roW[e][1], base + 51, nullptr, nullptr, llrAll + (size_t)e * PTOT * 4, ro, 4, 4, 1, 0);
        conv(hbuf,  roW[e][2], base + 53, nullptr, robuf, nullptr, 64, ro, ro, 1, 1);
        conv(robuf, roW[e][3], base + 55, nullptr, nullptr, chanAll + (size_t)e * PTOT * 8, ro, 8, 8, 1, 0);
    }

    size_t ncomb = (size_t)64 * 12 * PIXB;
    combine_kernel<<<dim3((unsigned)((ncomb + 255) / 256)), 256, 0, stream>>>(
        llrAll, chanAll, sel, outLogits, outChan);
}
